// MinigridStateSequenceNet_12824772346593
// MI455X (gfx1250) — matrix operand order repaired and checked by reading, not compiled, not run on hardware
//
#include <hip/hip_runtime.h>
#include <hip/hip_bf16.h>

// ---------------------------------------------------------------------------
// MinigridStateSequenceNet on MI455X (gfx1250, wave32, WMMA f16->f32)
// ---------------------------------------------------------------------------
#define T_DIM   256
#define B_DIM   64
#define HIST    16
#define HID     128
#define EMB     128
#define NSEQ    (T_DIM * B_DIM)          // 16384
#define GATES   (4 * HID)                // 512
#define PADROWS ((T_DIM + HIST - 1) * B_DIM)  // 17344

typedef __attribute__((ext_vector_type(16))) _Float16 v16h;
typedef __attribute__((ext_vector_type(8)))  _Float16 v8h;
typedef __attribute__((ext_vector_type(8)))  float    v8f;
typedef __attribute__((ext_vector_type(4)))  float    v4f;

__device__ __forceinline__ float elu1(float x)  { return x > 0.f ? x : (__expf(x) - 1.f); }
__device__ __forceinline__ float sigm(float x)  { return 1.f / (1.f + __expf(-x)); }

__device__ __forceinline__ v16h frag_cat(v8h lo, v8h hi) {
    return __builtin_shufflevector(lo, hi, 0,1,2,3,4,5,6,7,8,9,10,11,12,13,14,15);
}

// 16-bit A-matrix 16x32 fragment (ISA 7.12.2): lanes 0-15 row M=lane, K 0-7 then 16-23;
// lanes 16-31 row M=lane-16, K 8-15 then 24-31. Row-major [rows x 128] f16 source.
__device__ __forceinline__ v16h load_A(const _Float16* base, int kc, int lane) {
    int m = lane & 15, half = lane >> 4;
    const _Float16* r = base + m * 128 + kc * 32 + half * 8;
    v8h lo = *reinterpret_cast<const v8h*>(r);
    v8h hi = *reinterpret_cast<const v8h*>(r + 16);
    return frag_cat(lo, hi);
}

// 16-bit B-matrix 32x16 fragment: B[k,n] = W[n,k], W row-major [N x 128] f16.
// lanes 0-15 hold K 0-7 then 16-23, lanes 16-31 hold K 8-15 then 24-31 for column n = passed per-lane.
__device__ __forceinline__ v16h load_B(const _Float16* W, int n, int kc, int lane) {
    int khalf = lane >> 4;
    const _Float16* r = W + n * 128 + kc * 32 + khalf * 8;
    v8h lo = *reinterpret_cast<const v8h*>(r);
    v8h hi = *reinterpret_cast<const v8h*>(r + 16);
    return frag_cat(lo, hi);
}

__device__ __forceinline__ v8f wmma_f16(v16h a, v16h b, v8f c) {
    return __builtin_amdgcn_wmma_f32_16x16x32_f16(false, a, false, b, (short)0, c, false, false);
}

// ---------------------------------------------------------------------------
// Kernel 0: fold readin into w_ih (no nonlinearity between them in the ref):
//   eff_W = w_ih @ readin_w  (512x128),  eff_b = w_ih @ readin_b + b_ih + b_hh
// plus f16 conversions of w_hh / readout_w.
// ---------------------------------------------------------------------------
__global__ void __launch_bounds__(256)
prep_eff(const float* __restrict__ w_ih, const float* __restrict__ readin_w,
         const float* __restrict__ readin_b, const float* __restrict__ b_ih,
         const float* __restrict__ b_hh, const float* __restrict__ w_hh,
         const float* __restrict__ readout_w,
         _Float16* __restrict__ effW16, float* __restrict__ effb,
         _Float16* __restrict__ whh16, _Float16* __restrict__ row16) {
    int idx = blockIdx.x * 256 + threadIdx.x;   // 0 .. 512*128-1
    int n = idx >> 7, k = idx & 127;
    float acc = 0.f;
    for (int j = 0; j < 128; ++j)
        acc += w_ih[n * 128 + j] * readin_w[j * 128 + k];
    effW16[idx] = (_Float16)acc;
    whh16[idx]  = (_Float16)w_hh[idx];
    if (n < 128) row16[idx] = (_Float16)readout_w[idx];
    if (k == 0) {
        float b = b_ih[n] + b_hh[n];
        for (int j = 0; j < 128; ++j)
            b += w_ih[n * 128 + j] * readin_b[j];
        effb[n] = b;
    }
}

// Padded history rows (x == 0) carry pre-gates == eff_b. Column-major pre_t.
__global__ void __launch_bounds__(256)
fill_bias(const float* __restrict__ effb, _Float16* __restrict__ pre_t) {
    int i = blockIdx.x * 256 + threadIdx.x;     // < GATES * (HIST-1)*B
    int n = i / ((HIST - 1) * B_DIM);
    int row = i - n * ((HIST - 1) * B_DIM);
    pre_t[(size_t)n * PADROWS + row] = (_Float16)effb[n];
}

// ---------------------------------------------------------------------------
// Kernel 1: conv stack. One wave32 per image, 8 images per block.
// Ref transpose (0,3,2,1): x[n,c,a,b] = inputs[n, h=b, w=a, c].
// ---------------------------------------------------------------------------
__global__ void __launch_bounds__(256)
conv_embed(const float* __restrict__ inp,
           const float* __restrict__ w1, const float* __restrict__ b1,
           const float* __restrict__ w2, const float* __restrict__ b2,
           const float* __restrict__ w3, const float* __restrict__ b3,
           _Float16* __restrict__ emb) {
    __shared__ float in_s[8][152];
    __shared__ float o1_s[8][512];
    __shared__ float o2_s[8][128];
    int wave = threadIdx.x >> 5, lane = threadIdx.x & 31;
    int img = blockIdx.x * 8 + wave;
    const float* x = inp + img * 147;

    for (int idx = lane; idx < 147; idx += 32) {
        int c = idx / 49, rem = idx - c * 49, a = rem / 7, b = rem - a * 7;
        in_s[wave][idx] = x[b * 21 + a * 3 + c];   // layout c*49 + a*7 + b
    }
    __syncthreads();

    // conv1: 3 -> 32, 7x7 -> 4x4, stride 2 pad 1
    {
        int o = lane;
        float bias = b1[o];
        for (int pq = 0; pq < 16; ++pq) {
            int p = pq >> 2, q = pq & 3;
            float acc = bias;
            for (int i = 0; i < 3; ++i)
                for (int kh = 0; kh < 3; ++kh) {
                    int a = 2 * p - 1 + kh;
                    if ((unsigned)a > 6u) continue;
                    for (int kw = 0; kw < 3; ++kw) {
                        int b = 2 * q - 1 + kw;
                        if ((unsigned)b > 6u) continue;
                        acc += in_s[wave][i * 49 + a * 7 + b] *
                               w1[((o * 3 + i) * 3 + kh) * 3 + kw];
                    }
                }
            o1_s[wave][(o << 4) + pq] = elu1(acc);
        }
    }
    __syncthreads();

    // conv2: 32 -> 32, 4x4 -> 2x2
    {
        int o = lane;
        float bias = b2[o];
        for (int pq = 0; pq < 4; ++pq) {
            int p = pq >> 1, q = pq & 1;
            float acc = bias;
            for (int i = 0; i < 32; ++i)
                for (int kh = 0; kh < 3; ++kh) {
                    int a = 2 * p - 1 + kh;
                    if ((unsigned)a > 3u) continue;
                    for (int kw = 0; kw < 3; ++kw) {
                        int b = 2 * q - 1 + kw;
                        if ((unsigned)b > 3u) continue;
                        acc += o1_s[wave][(i << 4) + (a << 2) + b] *
                               w2[((o * 32 + i) * 3 + kh) * 3 + kw];
                    }
                }
            o2_s[wave][(o << 2) + pq] = elu1(acc);
        }
    }
    __syncthreads();

    // conv3: 32 -> 128, 2x2 -> 1x1 (only taps kh,kw in {1,2} hit valid input)
    for (int r = 0; r < 4; ++r) {
        int o = lane + (r << 5);
        float acc = b3[o];
        for (int i = 0; i < 32; ++i)
            for (int kh = 1; kh < 3; ++kh)
                for (int kw = 1; kw < 3; ++kw)
                    acc += o2_s[wave][(i << 2) + ((kh - 1) << 1) + (kw - 1)] *
                           w3[((o * 32 + i) * 3 + kh) * 3 + kw];
        emb[img * 128 + o] = (_Float16)elu1(acc);
    }
}

// ---------------------------------------------------------------------------
// Kernel 2: pre-gates GEMM  (16384 x 128) x (128 x 512) + eff_b.
// Output stored COLUMN-MAJOR pre_t[gate][padrow] so the LSTM kernel can seed
// its D-layout accumulators with b128 loads (lane = fixed N, contiguous M).
// ---------------------------------------------------------------------------
__global__ void __launch_bounds__(256)
pre_gemm(const _Float16* __restrict__ emb, const _Float16* __restrict__ effW,
         const float* __restrict__ effb, _Float16* __restrict__ pre_t) {
    int wave = threadIdx.x >> 5, lane = threadIdx.x & 31;
    int gw = blockIdx.x * 8 + wave;
    int mbase = (gw >> 3) * 16;
    int nbase = (gw & 7) * 64;
    int half = lane >> 4;

    v8f acc[4];
#pragma unroll
    for (int tn = 0; tn < 4; ++tn) {
        float b = effb[nbase + tn * 16 + (lane & 15)];
#pragma unroll
        for (int r = 0; r < 8; ++r) acc[tn][r] = b;
    }
#pragma unroll
    for (int kc = 0; kc < 4; ++kc) {
        v16h a = load_A(emb + mbase * 128, kc, lane);
#pragma unroll
        for (int tn = 0; tn < 4; ++tn) {
            v16h bm = load_B(effW, nbase + tn * 16 + (lane & 15), kc, lane);
            acc[tn] = wmma_f16(a, bm, acc[tn]);
        }
    }
    // D layout: lane -> fixed n, VGPR r -> M = mbase + r + half*8.
    // Transposed rows are contiguous -> one packed b128 store per tile.
#pragma unroll
    for (int tn = 0; tn < 4; ++tn) {
        int n = nbase + tn * 16 + (lane & 15);
        v8h pk;
#pragma unroll
        for (int r = 0; r < 8; ++r) pk[r] = (_Float16)acc[tn][r];
        *reinterpret_cast<v8h*>(pre_t + (size_t)n * PADROWS +
                                mbase + (HIST - 1) * B_DIM + half * 8) = pk;
    }
}

// ---------------------------------------------------------------------------
// Kernel 3: 16-step LSTM over each length-16 history window + fused readout.
// Block = 16 sequences, 8 waves; wave owns a 64-wide gate strip.
// w_hh fragments stay resident in VGPRs across all steps; h-state in LDS (f16).
// ---------------------------------------------------------------------------
__global__ void __launch_bounds__(256)
lstm_readout(const _Float16* __restrict__ pre_t, const _Float16* __restrict__ whh16,
             const _Float16* __restrict__ row16, const float* __restrict__ readout_b,
             const unsigned char* __restrict__ done, float* __restrict__ out) {
    __shared__ __align__(16) _Float16 h_s[16 * 128];
    __shared__ float c_s[16 * 128];
    __shared__ __align__(16) float g_t[512 * 16];   // transposed gates [n][m]

    int tid = threadIdx.x;
    int wave = tid >> 5, lane = tid & 31, half = lane >> 4;
    int mbase = blockIdx.x * 16;
    int nbase = wave * 64;

    // resident w_hh B fragments: 4 N-tiles x 4 K-chunks = 128 VGPRs
    v16h Bf[16];
#pragma unroll
    for (int tn = 0; tn < 4; ++tn)
#pragma unroll
        for (int kc = 0; kc < 4; ++kc)
            Bf[tn * 4 + kc] = load_B(whh16, nbase + tn * 16 + (lane & 15), kc, lane);

    int m0 = tid >> 4;            // 0..15 : sequence-in-block for cell math
    int j0 = (tid & 15) * 8;      // 0..120: hidden slice for cell math
    int s_m = mbase + m0;         // global sequence index
#pragma unroll
    for (int q = 0; q < 8; ++q) {
        h_s[m0 * 128 + j0 + q] = (_Float16)0.f;
        c_s[m0 * 128 + j0 + q] = 0.f;
    }
    __syncthreads();

    for (int hstep = 0; hstep < HIST; ++hstep) {
        // done-mask reset (no-op at hstep==0 since state is zero, matching ref)
        int didx = s_m + (hstep - (HIST - 1)) * B_DIM;
        if (didx >= 0 && done[didx]) {
#pragma unroll
            for (int q = 0; q < 8; ++q) {
                h_s[m0 * 128 + j0 + q] = (_Float16)0.f;
                c_s[m0 * 128 + j0 + q] = 0.f;
            }
        }
        __syncthreads();

        // gates = pre[step] + h @ w_hh^T. Seed from column-major pre_t:
        // padrow(s,h) = s + h*B, rows contiguous per lane -> b128 loads.
        v8f acc[4];
#pragma unroll
        for (int tn = 0; tn < 4; ++tn) {
            int n = nbase + tn * 16 + (lane & 15);
            const _Float16* p =
                pre_t + (size_t)n * PADROWS + mbase + hstep * B_DIM + half * 8;
            v8h pk = *reinterpret_cast<const v8h*>(p);
#pragma unroll
            for (int r = 0; r < 8; ++r) acc[tn][r] = (float)pk[r];
            if (hstep + 1 < HIST)
                __builtin_prefetch(p + B_DIM, 0, 3);  // global_prefetch_b8
        }
#pragma unroll
        for (int kc = 0; kc < 4; ++kc) {
            v16h a = load_A(h_s, kc, lane);
#pragma unroll
            for (int tn = 0; tn < 4; ++tn)
                acc[tn] = wmma_f16(a, Bf[tn * 4 + kc], acc[tn]);
        }
        // spill gates transposed: lane's 8 rows contiguous -> 2x b128 per tile
#pragma unroll
        for (int tn = 0; tn < 4; ++tn) {
            int n = nbase + tn * 16 + (lane & 15);
            v4f lo = { acc[tn][0], acc[tn][1], acc[tn][2], acc[tn][3] };
            v4f hi = { acc[tn][4], acc[tn][5], acc[tn][6], acc[tn][7] };
            *reinterpret_cast<v4f*>(g_t + n * 16 + half * 8)     = lo;
            *reinterpret_cast<v4f*>(g_t + n * 16 + half * 8 + 4) = hi;
        }
        __syncthreads();

        // LSTM cell math, 8 (m,j) pairs per thread (gates read transposed)
#pragma unroll
        for (int q = 0; q < 8; ++q) {
            int j = j0 + q;
            float ig = g_t[j * 16 + m0];
            float fg = g_t[(128 + j) * 16 + m0];
            float gg = g_t[(256 + j) * 16 + m0];
            float og = g_t[(384 + j) * 16 + m0];
            float c2 = sigm(fg) * c_s[m0 * 128 + j] + sigm(ig) * tanhf(gg);
            float h2 = sigm(og) * tanhf(c2);
            c_s[m0 * 128 + j] = c2;
            h_s[m0 * 128 + j] = (_Float16)h2;
        }
        __syncthreads();
    }

    // fused readout: out = h_final @ readout_w^T + readout_b (wave w -> N-tile w)
    int nb = wave * 16;
    v8f acc;
    {
        float b = readout_b[nb + (lane & 15)];
#pragma unroll
        for (int r = 0; r < 8; ++r) acc[r] = b;
    }
#pragma unroll
    for (int kc = 0; kc < 4; ++kc) {
        v16h a  = load_A(h_s, kc, lane);
        v16h bm = load_B(row16, nb + (lane & 15), kc, lane);
        acc = wmma_f16(a, bm, acc);
    }
#pragma unroll
    for (int r = 0; r < 8; ++r) {
        int s = mbase + r + half * 8;
        out[s * 128 + nb + (lane & 15)] = acc[r];
    }
}

// ---------------------------------------------------------------------------
// Workspace layout (bytes)
// ---------------------------------------------------------------------------
static constexpr size_t OFF_EMB  = 0;                                  // f16 16384x128
static constexpr size_t SZ_EMB   = (size_t)NSEQ * 128 * 2;
static constexpr size_t OFF_PRE  = OFF_EMB + SZ_EMB;                   // f16 512 x 17344 (col-major)
static constexpr size_t SZ_PRE   = (size_t)PADROWS * GATES * 2;
static constexpr size_t OFF_EFFW = OFF_PRE + SZ_PRE;                   // f16 512x128
static constexpr size_t OFF_WHH  = OFF_EFFW + (size_t)GATES * 128 * 2; // f16 512x128
static constexpr size_t OFF_ROW  = OFF_WHH + (size_t)GATES * 128 * 2;  // f16 128x128
static constexpr size_t OFF_EFFB = OFF_ROW + (size_t)128 * 128 * 2;    // f32 512

extern "C" void kernel_launch(void* const* d_in, const int* in_sizes, int n_in,
                              void* d_out, int out_size, void* d_ws, size_t ws_size,
                              hipStream_t stream) {
    const float* inputs    = (const float*)d_in[0];
    const unsigned char* done = (const unsigned char*)d_in[1];
    const float* conv1_w   = (const float*)d_in[2];
    const float* conv1_b   = (const float*)d_in[3];
    const float* conv2_w   = (const float*)d_in[4];
    const float* conv2_b   = (const float*)d_in[5];
    const float* conv3_w   = (const float*)d_in[6];
    const float* conv3_b   = (const float*)d_in[7];
    const float* readin_w  = (const float*)d_in[8];
    const float* readin_b  = (const float*)d_in[9];
    const float* w_ih      = (const float*)d_in[10];
    const float* w_hh      = (const float*)d_in[11];
    const float* b_ih      = (const float*)d_in[12];
    const float* b_hh      = (const float*)d_in[13];
    const float* readout_w = (const float*)d_in[14];
    const float* readout_b = (const float*)d_in[15];
    float* out = (float*)d_out;

    char* ws = (char*)d_ws;
    _Float16* emb    = (_Float16*)(ws + OFF_EMB);
    _Float16* pre_t  = (_Float16*)(ws + OFF_PRE);
    _Float16* effW16 = (_Float16*)(ws + OFF_EFFW);
    _Float16* whh16  = (_Float16*)(ws + OFF_WHH);
    _Float16* row16  = (_Float16*)(ws + OFF_ROW);
    float*    effb   = (float*)(ws + OFF_EFFB);

    // in-order stream => dependencies satisfied by launch order
    prep_eff<<<(GATES * 128) / 256, 256, 0, stream>>>(
        w_ih, readin_w, readin_b, b_ih, b_hh, w_hh, readout_w,
        effW16, effb, whh16, row16);

    fill_bias<<<(GATES * (HIST - 1) * B_DIM) / 256, 256, 0, stream>>>(effb, pre_t);

    conv_embed<<<NSEQ / 8, 256, 0, stream>>>(
        inputs, conv1_w, conv1_b, conv2_w, conv2_b, conv3_w, conv3_b, emb);

    pre_gemm<<<(NSEQ / 16) * (GATES / 64) / 8, 256, 0, stream>>>(emb, effW16, effb, pre_t);

    lstm_readout<<<NSEQ / 16, 256, 0, stream>>>(pre_t, whh16, row16, readout_b, done, out);
}
